// MultiLayerTransformer_67327907332639
// MI455X (gfx1250) — matrix operand order repaired and checked by reading, not compiled, not run on hardware
//
#include <hip/hip_runtime.h>
#include <hip/hip_bf16.h>

// ---------------------------------------------------------------------------
// Problem constants (match reference)
// ---------------------------------------------------------------------------
#define NN   160000      // nodes
#define NE   320000      // hyperedges
#define NI   1600000     // incidences
#define NC   800         // cells (= 100*8)
#define DH   64
#define DIN  768
#define IJ   800         // 100*8
#define BN_EPS 1e-5f

typedef __attribute__((ext_vector_type(16))) _Float16 v16h;
typedef __attribute__((ext_vector_type(8)))  float    v8f;

union V16H { v16h v; _Float16 h[16]; };

// ---------------------------------------------------------------------------
// Helpers
// ---------------------------------------------------------------------------
__device__ __forceinline__ int lower_bound_i(const int* __restrict__ a, int n, int key) {
    int lo = 0, hi = n;
    while (lo < hi) {
        int mid = (lo + hi) >> 1;
        if (a[mid] < key) lo = mid + 1; else hi = mid;
    }
    return lo;
}

// ---------------------------------------------------------------------------
// Pre-swizzle a [K,64] f32 row-major weight matrix into WMMA B-fragment order
// (f16).  Fragment index layout: wf[((t*4 + j)*32 + lane)*16 + e]
//   t = K-step (32 K per step), j = 16-col tile, lane: 0-15 -> N=lane,
//   K=e (e<8) / 8+e (e>=8) ; 16-31 -> N=lane-16, K=8+e (e<8) / 16+e (e>=8)
// ---------------------------------------------------------------------------
__global__ void prep_wfrag(const float* __restrict__ W, int K, _Float16* __restrict__ wf) {
    int gid = blockIdx.x * blockDim.x + threadIdx.x;
    int total = (K >> 5) * 4 * 32 * 16;
    if (gid >= total) return;
    int e = gid & 15;
    int l = (gid >> 4) & 31;
    int j = (gid >> 9) & 3;
    int t = gid >> 11;
    int half = l >> 4;
    int n = (l & 15) + j * 16;
    int k = t * 32 + half * 8 + (e & 7) + ((e >> 3) << 4);
    wf[gid] = (_Float16)W[k * 64 + n];
}

// ---------------------------------------------------------------------------
// WMMA GEMM: Y[M,64] = gather(A, idx)[M,K] @ W[K,64] (+ bias)
// One wave -> 16 rows x 64 cols (4 accumulators).  K multiple of 32,
// M multiple of 16.  idx == nullptr -> identity gather.
// ---------------------------------------------------------------------------
__global__ __launch_bounds__(256) void gemm_wmma(
    const float* __restrict__ A, const int* __restrict__ idx,
    int M, int K, const _Float16* __restrict__ wf,
    const float* __restrict__ bias, float* __restrict__ Y)
{
    const int wave = threadIdx.x >> 5;
    const int lane = threadIdx.x & 31;
    const int rowBase = blockIdx.x * 128 + wave * 16;
    if (rowBase >= M) return;                 // wave-uniform: EXEC stays all-1s

    const int half = lane >> 4;
    const int nlo  = lane & 15;

    const int row    = rowBase + nlo;         // A tile row handled by this lane
    const int srcRow = idx ? idx[row] : row;
    const float* Ab  = A + (size_t)srcRow * K + half * 8;

    const int ksteps = K >> 5;
    v8f acc[4] = {};

    for (int t = 0; t < ksteps; ++t) {
        const float* Ap = Ab + t * 32;
        float4 q0 = *(const float4*)(Ap);
        float4 q1 = *(const float4*)(Ap + 4);
        float4 q2 = *(const float4*)(Ap + 16);
        float4 q3 = *(const float4*)(Ap + 20);
        V16H a;
        a.h[0]=(_Float16)q0.x; a.h[1]=(_Float16)q0.y; a.h[2]=(_Float16)q0.z; a.h[3]=(_Float16)q0.w;
        a.h[4]=(_Float16)q1.x; a.h[5]=(_Float16)q1.y; a.h[6]=(_Float16)q1.z; a.h[7]=(_Float16)q1.w;
        a.h[8]=(_Float16)q2.x; a.h[9]=(_Float16)q2.y; a.h[10]=(_Float16)q2.z; a.h[11]=(_Float16)q2.w;
        a.h[12]=(_Float16)q3.x; a.h[13]=(_Float16)q3.y; a.h[14]=(_Float16)q3.z; a.h[15]=(_Float16)q3.w;

        const _Float16* wt = wf + (size_t)t * 2048 + lane * 16;
        #pragma unroll
        for (int j = 0; j < 4; ++j) {
            v16h b = *(const v16h*)(wt + j * 512);
            acc[j] = __builtin_amdgcn_wmma_f32_16x16x32_f16(
                false, a.v, false, b, (short)0, acc[j], false, false);
        }
    }

    #pragma unroll
    for (int j = 0; j < 4; ++j) {
        const int n  = j * 16 + nlo;
        const float bi = bias ? bias[n] : 0.0f;
        #pragma unroll
        for (int r = 0; r < 8; ++r) {
            const int mm = r + half * 8;                  // C layout: VGPR r -> M=r / 8+r
            Y[(size_t)(rowBase + mm) * 64 + n] = acc[j][r] + bi;
        }
    }
}

// ---------------------------------------------------------------------------
// Degree counting + inverse-safe
// ---------------------------------------------------------------------------
__global__ void count_deg(const int* __restrict__ inc_node, const int* __restrict__ inc_hedge,
                          int n, unsigned* __restrict__ cnt_v, unsigned* __restrict__ cnt_e)
{
    int gid = blockIdx.x * blockDim.x + threadIdx.x;
    if (gid >= n) return;
    atomicAdd(&cnt_e[inc_hedge[gid]], 1u);
    atomicAdd(&cnt_v[inc_node[gid]], 1u);
}

__global__ void inv_count(const unsigned* __restrict__ cnt, float* __restrict__ inv, int n) {
    int gid = blockIdx.x * blockDim.x + threadIdx.x;
    if (gid >= n) return;
    unsigned c = cnt[gid];
    inv[gid] = c > 0 ? 1.0f / (float)c : 0.0f;
}

// ---------------------------------------------------------------------------
// Deterministic hedge reduction (inc_hedge is sorted): wave per hyperedge.
// he[e,:] = (sum over members of xt[inc_node[i],:]) * inv_deg_e[e]
// ---------------------------------------------------------------------------
__global__ void hedge_reduce(const int* __restrict__ inc_hedge, const int* __restrict__ inc_node,
                             const float* __restrict__ xt, const float* __restrict__ inv_deg_e,
                             float* __restrict__ he, int ni, int nhe)
{
    int gtid = blockIdx.x * blockDim.x + threadIdx.x;
    int e    = gtid >> 5;
    int lane = gtid & 31;
    if (e >= nhe) return;
    int lo = lower_bound_i(inc_hedge, ni, e);
    int hi = lower_bound_i(inc_hedge, ni, e + 1);
    int f = lane * 2;
    float s0 = 0.0f, s1 = 0.0f;
    for (int i = lo; i < hi; ++i) {
        int v = inc_node[i];
        float2 q = *(const float2*)(xt + (size_t)v * 64 + f);
        s0 += q.x; s1 += q.y;
    }
    float inv = inv_deg_e[e];
    he[(size_t)e * 64 + f]     = s0 * inv;
    he[(size_t)e * 64 + f + 1] = s1 * inv;
}

// ---------------------------------------------------------------------------
// Node scatter (inc_node unsorted): 16 threads per incidence, 4 feats each.
// ---------------------------------------------------------------------------
__global__ void node_scatter(const int* __restrict__ inc_node, const int* __restrict__ inc_hedge,
                             const float* __restrict__ he, float* __restrict__ acc, int ni)
{
    int gid = blockIdx.x * blockDim.x + threadIdx.x;
    if (gid >= ni * 16) return;
    int i = gid >> 4;
    int c = (gid & 15) * 4;
    int v = inc_node[i];
    int h = inc_hedge[i];
    float4 q = *(const float4*)(he + (size_t)h * 64 + c);
    float* dst = acc + (size_t)v * 64 + c;
    atomicAdd(dst + 0, q.x);
    atomicAdd(dst + 1, q.y);
    atomicAdd(dst + 2, q.z);
    atomicAdd(dst + 3, q.w);
}

// out[i] = relu(acc[i]*inv_deg_v[row] + bias[col]) + nf[i]
__global__ void node_finish(const float* __restrict__ acc, const float* __restrict__ invv,
                            const float* __restrict__ bias, const float* __restrict__ nf,
                            float* __restrict__ out, int m)
{
    int gid = blockIdx.x * blockDim.x + threadIdx.x;
    if (gid >= m * 64) return;
    int row = gid >> 6, col = gid & 63;
    float v = acc[gid] * invv[row] + bias[col];
    out[gid] = fmaxf(v, 0.0f) + nf[gid];
}

// ---------------------------------------------------------------------------
// Cell pooling (node2cell sorted): wave per cell, mean+max, 2 feats/lane.
// ---------------------------------------------------------------------------
__global__ void pool_cells(const int* __restrict__ node2cell, const float* __restrict__ gx,
                           float* __restrict__ cell, int nn, int nc)
{
    int gtid = blockIdx.x * blockDim.x + threadIdx.x;
    int c    = gtid >> 5;
    int lane = gtid & 31;
    if (c >= nc) return;
    int lo = lower_bound_i(node2cell, nn, c);
    int hi = lower_bound_i(node2cell, nn, c + 1);
    int f = lane * 2;
    float s0 = 0.0f, s1 = 0.0f;
    float m0 = -3.402823466e38f, m1 = -3.402823466e38f;
    for (int i = lo; i < hi; ++i) {
        float2 q = *(const float2*)(gx + (size_t)i * 64 + f);
        s0 += q.x; s1 += q.y;
        m0 = fmaxf(m0, q.x); m1 = fmaxf(m1, q.y);
    }
    int cnt = hi - lo;
    float inv = cnt > 0 ? 1.0f / (float)cnt : 0.0f;
    float r0 = cnt > 0 ? 0.5f * (s0 * inv + m0) : 0.0f;
    float r1 = cnt > 0 ? 0.5f * (s1 * inv + m1) : 0.0f;
    cell[(size_t)c * 64 + f]     = r0;
    cell[(size_t)c * 64 + f + 1] = r1;
}

// x_upd = x0 + 0.3 * cell_emb  (written straight into d_out)
__global__ void xupd_kernel(const float* __restrict__ x0, const float* __restrict__ cell,
                            float* __restrict__ out, int n)
{
    int gid = blockIdx.x * blockDim.x + threadIdx.x;
    if (gid >= n) return;
    out[gid] = x0[gid] + 0.3f * cell[gid];
}

// ---------------------------------------------------------------------------
// TCN branch: block per sample (i), thread per channel (o).
// ---------------------------------------------------------------------------
__global__ __launch_bounds__(64) void tcn_kernel(
    const float* __restrict__ xupd,
    const float* wc1, const float* bc1, const float* g1, const float* be1, const float* m1, const float* v1,
    const float* wc2, const float* bc2, const float* g2, const float* be2, const float* m2, const float* v2,
    const float* wc3, const float* bc3, const float* g3, const float* be3, const float* m3, const float* v3,
    const float* __restrict__ fct_w, const float* __restrict__ fct_b, float* __restrict__ out2)
{
    __shared__ float bufA[64][8];
    __shared__ float bufB[64][8];
    __shared__ float mv[64];

    const int i = blockIdx.x;
    const int o = threadIdx.x;

    // h[i, c, t] = x_upd[i, t, c]
    for (int t = 0; t < 8; ++t) bufA[o][t] = xupd[i * 512 + t * 64 + o];
    __syncthreads();

    float (*a)[8] = bufA;
    float (*b)[8] = bufB;
    const float* WC[3] = {wc1, wc2, wc3};
    const float* BC[3] = {bc1, bc2, bc3};
    const float* GG[3] = {g1, g2, g3};
    const float* BE[3] = {be1, be2, be3};
    const float* MM[3] = {m1, m2, m3};
    const float* VV[3] = {v1, v2, v3};

    for (int L = 0; L < 3; ++L) {
        const float* w = WC[L] + o * 192;      // wc[o][c][k], OIH layout
        const float bias  = BC[L][o];
        const float scale = GG[L][o] * rsqrtf(VV[L][o] + BN_EPS);
        const float mean  = MM[L][o];
        const float beta  = BE[L][o];
        float res[8];
        #pragma unroll
        for (int t = 0; t < 8; ++t) res[t] = bias;
        for (int c = 0; c < 64; ++c) {
            float w0 = w[c * 3 + 0], w1 = w[c * 3 + 1], w2 = w[c * 3 + 2];
            float hc[8];
            #pragma unroll
            for (int t = 0; t < 8; ++t) hc[t] = a[c][t];
            #pragma unroll
            for (int t = 0; t < 8; ++t) {
                float s = hc[t] * w1;
                if (t > 0) s += hc[t - 1] * w0;
                if (t < 7) s += hc[t + 1] * w2;
                res[t] += s;
            }
        }
        #pragma unroll
        for (int t = 0; t < 8; ++t) {
            float s = fmaxf(res[t], 0.0f);
            b[o][t] = (s - mean) * scale + beta;
        }
        __syncthreads();
        float (*tmp)[8] = a; a = b; b = tmp;
    }

    float s = 0.0f;
    #pragma unroll
    for (int t = 0; t < 8; ++t) s += a[o][t];
    mv[o] = s * 0.125f;
    __syncthreads();

    float r = fct_b[o];
    for (int c = 0; c < 64; ++c) r += mv[c] * fct_w[c * 64 + o];
    out2[i * 64 + o] = r;
}

// ---------------------------------------------------------------------------
// Host launcher
// ---------------------------------------------------------------------------
extern "C" void kernel_launch(void* const* d_in, const int* in_sizes, int n_in,
                              void* d_out, int out_size, void* d_ws, size_t ws_size,
                              hipStream_t stream)
{
    (void)in_sizes; (void)n_in; (void)out_size; (void)ws_size;

    const float* x           = (const float*)d_in[0];
    const int*   node_tokens = (const int*)  d_in[1];
    const int*   inc_node    = (const int*)  d_in[2];
    const int*   inc_hedge   = (const int*)  d_in[3];
    const int*   node2cell   = (const int*)  d_in[4];
    const float* vac_emb     = (const float*)d_in[5];
    const float* fhi_w = (const float*)d_in[6];  const float* fhi_b = (const float*)d_in[7];
    const float* fin_w = (const float*)d_in[8];  const float* fin_b = (const float*)d_in[9];
    const float* th1   = (const float*)d_in[10]; const float* b1    = (const float*)d_in[11];
    const float* th2   = (const float*)d_in[12]; const float* b2    = (const float*)d_in[13];
    const float* wc1 = (const float*)d_in[14]; const float* bc1 = (const float*)d_in[15];
    const float* g1  = (const float*)d_in[16]; const float* be1 = (const float*)d_in[17];
    const float* m1  = (const float*)d_in[18]; const float* v1  = (const float*)d_in[19];
    const float* wc2 = (const float*)d_in[20]; const float* bc2 = (const float*)d_in[21];
    const float* g2  = (const float*)d_in[22]; const float* be2 = (const float*)d_in[23];
    const float* m2  = (const float*)d_in[24]; const float* v2  = (const float*)d_in[25];
    const float* wc3 = (const float*)d_in[26]; const float* bc3 = (const float*)d_in[27];
    const float* g3  = (const float*)d_in[28]; const float* be3 = (const float*)d_in[29];
    const float* m3  = (const float*)d_in[30]; const float* v3  = (const float*)d_in[31];
    const float* fct_w = (const float*)d_in[32]; const float* fct_b = (const float*)d_in[33];

    float* out = (float*)d_out;             // [0,51200) = x_upd ; [51200,57600) = out2

    // ---- workspace carve-up (256-B aligned) ----
    char* ws = (char*)d_ws;
    size_t off = 0;
    auto carve = [&](size_t bytes) -> char* {
        char* p = ws + off;
        off = (off + bytes + 255) & ~(size_t)255;
        return p;
    };
    float*    NF   = (float*)   carve((size_t)NN * DH * 4);
    float*    H    = (float*)   carve((size_t)NN * DH * 4);
    float*    XT   = (float*)   carve((size_t)NN * DH * 4);
    float*    ACC  = (float*)   carve((size_t)NN * DH * 4);
    float*    HE   = (float*)   carve((size_t)NE * DH * 4);
    float*    INVE = (float*)   carve((size_t)NE * 4);
    float*    INVV = (float*)   carve((size_t)NN * 4);
    unsigned* CNTE = (unsigned*)carve((size_t)NE * 4);
    unsigned* CNTV = (unsigned*)carve((size_t)NN * 4);
    float*    X0   = (float*)   carve((size_t)IJ * DH * 4);
    float*    CELL = (float*)   carve((size_t)IJ * DH * 4);
    _Float16* WFHI = (_Float16*)carve((size_t)(DIN / 32) * 2048 * 2);
    _Float16* WFIN = (_Float16*)carve((size_t)(DIN / 32) * 2048 * 2);
    _Float16* WT1  = (_Float16*)carve((size_t)(DH / 32) * 2048 * 2);
    _Float16* WT2  = (_Float16*)carve((size_t)(DH / 32) * 2048 * 2);

    // ---- zero accumulation buffers (every launch; replay-safe) ----
    hipMemsetAsync(CNTE, 0, (size_t)NE * 4, stream);
    hipMemsetAsync(CNTV, 0, (size_t)NN * 4, stream);
    hipMemsetAsync(ACC,  0, (size_t)NN * DH * 4, stream);

    // ---- degrees ----
    count_deg<<<(NI + 255) / 256, 256, 0, stream>>>(inc_node, inc_hedge, NI, CNTV, CNTE);
    inv_count<<<(NE + 255) / 256, 256, 0, stream>>>(CNTE, INVE, NE);
    inv_count<<<(NN + 255) / 256, 256, 0, stream>>>(CNTV, INVV, NN);

    // ---- weight fragment pre-swizzle (f32 -> f16, WMMA B layout) ----
    prep_wfrag<<<(DIN * 64 + 255) / 256, 256, 0, stream>>>(fhi_w, DIN, WFHI);
    prep_wfrag<<<(DIN * 64 + 255) / 256, 256, 0, stream>>>(fin_w, DIN, WFIN);
    prep_wfrag<<<(DH  * 64 + 255) / 256, 256, 0, stream>>>(th1,  DH,  WT1);
    prep_wfrag<<<(DH  * 64 + 255) / 256, 256, 0, stream>>>(th2,  DH,  WT2);

    // ---- nf = vac_emb[tokens] @ fhi_w + fhi_b  (gather GEMM, WMMA) ----
    gemm_wmma<<<(NN + 127) / 128, 256, 0, stream>>>(vac_emb, node_tokens, NN, DIN, WFHI, fhi_b, NF);
    // ---- x0 = x @ fin_w + fin_b ----
    gemm_wmma<<<(IJ + 127) / 128, 256, 0, stream>>>(x, nullptr, IJ, DIN, WFIN, fin_b, X0);

    // ---- hypergraph conv layer 1 ----
    gemm_wmma<<<(NN + 127) / 128, 256, 0, stream>>>(NF, nullptr, NN, DH, WT1, nullptr, XT);
    hedge_reduce<<<(NE * 32 + 255) / 256, 256, 0, stream>>>(inc_hedge, inc_node, XT, INVE, HE, NI, NE);
    node_scatter<<<(NI * 16 + 255) / 256, 256, 0, stream>>>(inc_node, inc_hedge, HE, ACC, NI);
    node_finish<<<(NN * 64 + 255) / 256, 256, 0, stream>>>(ACC, INVV, b1, NF, H, NN);

    // ---- hypergraph conv layer 2 ----
    hipMemsetAsync(ACC, 0, (size_t)NN * DH * 4, stream);
    gemm_wmma<<<(NN + 127) / 128, 256, 0, stream>>>(H, nullptr, NN, DH, WT2, nullptr, XT);
    hedge_reduce<<<(NE * 32 + 255) / 256, 256, 0, stream>>>(inc_hedge, inc_node, XT, INVE, HE, NI, NE);
    node_scatter<<<(NI * 16 + 255) / 256, 256, 0, stream>>>(inc_node, inc_hedge, HE, ACC, NI);
    node_finish<<<(NN * 64 + 255) / 256, 256, 0, stream>>>(ACC, INVV, b2, NF, H, NN);   // H = gx

    // ---- pooling + x update ----
    pool_cells<<<(NC * 32 + 255) / 256, 256, 0, stream>>>(node2cell, H, CELL, NN, NC);
    xupd_kernel<<<(IJ * DH + 255) / 256, 256, 0, stream>>>(X0, CELL, out, IJ * DH);

    // ---- TCN branch (reads x_upd from d_out, writes out2) ----
    tcn_kernel<<<100, 64, 0, stream>>>(out,
        wc1, bc1, g1, be1, m1, v1,
        wc2, bc2, g2, be2, m2, v2,
        wc3, bc3, g3, be3, m3, v3,
        fct_w, fct_b, out + 51200);
}
